// v8DetectionLoss_71287867179417
// MI455X (gfx1250) — matrix operand order repaired and checked by reading, not compiled, not run on hardware
//
#include <hip/hip_runtime.h>
#include <hip/hip_bf16.h>

#define DEV_INLINE __device__ __forceinline__

constexpr int B_  = 32;
constexpr int N_  = 8400;   // 80*80 + 40*40 + 20*20
constexpr int NC_ = 80;
constexpr int M_  = 40;
constexpr int TOPK_ = 10;

typedef __attribute__((ext_vector_type(16))) _Float16 v16h;
typedef __attribute__((ext_vector_type(8)))  float    v8f;

// ---------------------------------------------------------------- helpers ---

DEV_INLINE void anchor_xy(int n, float& ax, float& ay) {
  if (n < 6400)      { int r = n;        ax = ((r % 80) + 0.5f) * 8.f;  ay = ((r / 80) + 0.5f) * 8.f;  }
  else if (n < 8000) { int r = n - 6400; ax = ((r % 40) + 0.5f) * 16.f; ay = ((r / 40) + 0.5f) * 16.f; }
  else               { int r = n - 8000; ax = ((r % 20) + 0.5f) * 32.f; ay = ((r / 20) + 0.5f) * 32.f; }
}

// Must be bit-identical wherever it is recomputed (colmax equality filter),
// so fp-contract is disabled inside.
DEV_INLINE float align_val(float bx0, float by0, float bx1, float by1,
                           float gx0, float gy0, float gx1, float gy1,
                           float raw_score) {
#pragma clang fp contract(off)
  float iw = fminf(bx1, gx1) - fmaxf(bx0, gx0); iw = fmaxf(iw, 0.f);
  float ih = fminf(by1, gy1) - fmaxf(by0, gy0); ih = fmaxf(ih, 0.f);
  float inter = iw * ih;
  float a1 = (bx1 - bx0) * (by1 - by0);
  float a2 = (gx1 - gx0) * (gy1 - gy0);
  float iou = inter / (a1 + a2 - inter + 1e-16f);
  iou = fmaxf(iou, 0.f);
  float sc = 1.f / (1.f + __expf(-raw_score));
  float i2 = iou * iou;
  return sqrtf(sc) * (i2 * i2 * i2);   // score^0.5 * iou^6
}

DEV_INLINE float block_sum(float v, float* sbuf) {   // blockDim.x == 256
  int t = threadIdx.x;
  sbuf[t] = v; __syncthreads();
  for (int s = 128; s > 0; s >>= 1) {
    if (t < s) sbuf[t] += sbuf[t + s];
    __syncthreads();
  }
  float r = sbuf[0]; __syncthreads();
  return r;
}

// ---------------------------------------------------------------- kernels ---

__global__ void k_init(float* accum) {
  if (threadIdx.x < 4) accum[threadIdx.x] = 0.f;
}

// DFL: softmax(16) per (anchor,side), dot with dfl_weight via WMMA f16.
// Per wave: 16 rows = 4 anchors x 4 sides -> one v_wmma_f32_16x16x32_f16.
// A layout (16-bit A 16x32): lane<16 holds K=0..7, lane>=16 holds K=8..15 in
// halves 0..7; halves 8..15 (K=16..31) zero-padded.
// B layout (16-bit B 32x16): lane<16 holds K=0..7, lane>=16 holds K=8..15 (halves 0..7) broadcast
// to every column; halves 8..15 (K=16..31) zero. D columns are then identical.
__global__ __launch_bounds__(256) void k_dfl_bbox(const float* __restrict__ pd,
                                                  const float* __restrict__ dw,
                                                  float* __restrict__ pb) {
  const int lane  = threadIdx.x & 31;
  const int wave  = threadIdx.x >> 5;
  const int aBase = (blockIdx.x * 8 + wave) * 4;   // 4 anchors per wave (exact grid)
  const int row   = lane & 15;
  const int hv    = lane >> 4;                     // K-half selector
  const int aIdx  = aBase + (row >> 2);
  const int side  = row & 3;

  const float* src = pd + (size_t)aIdx * 64 + side * 16 + hv * 8;
  float4 p0 = *(const float4*)(src);
  float4 p1 = *(const float4*)(src + 4);
  float x[8] = {p0.x, p0.y, p0.z, p0.w, p1.x, p1.y, p1.z, p1.w};

  float mx = x[0];
#pragma unroll
  for (int i = 1; i < 8; i++) mx = fmaxf(mx, x[i]);
  mx = fmaxf(mx, __shfl_xor(mx, 16, 32));         // combine row halves
  float e[8], s = 0.f;
#pragma unroll
  for (int i = 0; i < 8; i++) { e[i] = __expf(x[i] - mx); s += e[i]; }
  s += __shfl_xor(s, 16, 32);
  float inv = 1.f / s;

  union { v16h v; _Float16 h[16]; } au, bu;
#pragma unroll
  for (int i = 0; i < 8; i++) { au.h[i] = (_Float16)(e[i] * inv); au.h[i + 8] = (_Float16)0.f; }
#pragma unroll
  for (int i = 0; i < 16; i++) bu.h[i] = (i < 8) ? (_Float16)dw[hv * 8 + i] : (_Float16)0.f;

  v8f c = {0.f, 0.f, 0.f, 0.f, 0.f, 0.f, 0.f, 0.f};
  v8f d = __builtin_amdgcn_wmma_f32_16x16x32_f16(false, au.v, false, bu.v,
                                                 (short)0, c, false, false);
  union { v8f v; float f[8]; } du; du.v = d;

  // D: lanes 0-15 VGPR r = row r; lanes 16-31 VGPR r = row r+8. Columns equal.
  if (row < 8) {
    // Static-index select chain (v_cndmask) instead of a runtime vector
    // subscript, which the compiler otherwise lowers via an LDS round-trip.
    float val = du.f[0];
#pragma unroll
    for (int i = 1; i < 8; i++) val = (row == i) ? du.f[i] : val;

    int r = row + hv * 8;                 // 16 active lanes -> 16 distinct rows
    int outA = aBase + (r >> 2);
    int sd = r & 3;                       // 0=l(x) 1=t(y) 2=r(x) 3=b(y)
    float ax, ay; anchor_xy(outA % N_, ax, ay);
    float coord = (sd & 1) ? ay : ax;
    pb[(size_t)outA * 4 + sd] = (sd < 2) ? (coord - val) : (coord + val);
  }
}

// Per anchor: column max of align over all 40 gts; also init atomicMin buffer.
__global__ __launch_bounds__(256) void k_colmax(const float* __restrict__ ps,
                                                const float* __restrict__ pb,
                                                const float* __restrict__ gtb,
                                                const int*   __restrict__ gtl,
                                                float* __restrict__ colmax,
                                                int*   __restrict__ tgtc) {
  int i = blockIdx.x * 256 + threadIdx.x;
  if (i >= B_ * N_) return;
  int b = i / N_;
  const float* bp = pb + (size_t)i * 4;
  float bx0 = bp[0], by0 = bp[1], bx1 = bp[2], by1 = bp[3];
  const float* g = gtb + (size_t)b * M_ * 4;
  const int*   l = gtl + b * M_;
  const float* srow = ps + (size_t)i * NC_;
  float mx = 0.f;                                  // align >= 0 always
  for (int m = 0; m < M_; m++) {
    float al = align_val(bx0, by0, bx1, by1,
                         g[m * 4], g[m * 4 + 1], g[m * 4 + 2], g[m * 4 + 3],
                         srow[l[m]]);
    mx = fmaxf(mx, al);
  }
  colmax[i] = mx;
  tgtc[i] = 0x7fffffff;
}

// One block per (b,m): align row in LDS, 10 sequential block-max passes
// (= exact top-10), then atomicMin assignment under the reference's filters.
__global__ __launch_bounds__(256) void k_assign(const float* __restrict__ ps,
                                                const float* __restrict__ pb,
                                                const float* __restrict__ gtb,
                                                const int*   __restrict__ gtl,
                                                const float* __restrict__ colmax,
                                                int* __restrict__ tgtc) {
  __shared__ float arow[N_];
  __shared__ float rv[256]; __shared__ int ri[256];
  __shared__ float selv[TOPK_]; __shared__ int seli[TOPK_];
  int bm = blockIdx.x;
  int b = bm / M_, m = bm % M_;
  const float* g = gtb + ((size_t)b * M_ + m) * 4;
  float gx0 = g[0], gy0 = g[1], gx1 = g[2], gy1 = g[3];
  int lbl = gtl[b * M_ + m];
  const float* psb = ps + (size_t)b * N_ * NC_;
  const float* pbb = pb + (size_t)b * N_ * 4;

  for (int n = threadIdx.x; n < N_; n += 256) {
    const float* bp = pbb + (size_t)n * 4;
    arow[n] = align_val(bp[0], bp[1], bp[2], bp[3], gx0, gy0, gx1, gy1,
                        psb[(size_t)n * NC_ + lbl]);
  }
  __syncthreads();

  for (int pass = 0; pass < TOPK_; pass++) {
    float bv = -1.f; int bi = 0;
    for (int n = threadIdx.x; n < N_; n += 256) {
      float v = arow[n];
      if (v > bv) { bv = v; bi = n; }
    }
    rv[threadIdx.x] = bv; ri[threadIdx.x] = bi; __syncthreads();
    for (int s = 128; s > 0; s >>= 1) {
      if (threadIdx.x < s && rv[threadIdx.x + s] > rv[threadIdx.x]) {
        rv[threadIdx.x] = rv[threadIdx.x + s]; ri[threadIdx.x] = ri[threadIdx.x + s];
      }
      __syncthreads();
    }
    if (threadIdx.x == 0) {
      selv[pass] = rv[0]; seli[pass] = ri[0];
      arow[ri[0]] = -1.f;                 // exclude from later passes
    }
    __syncthreads();
  }

  if (threadIdx.x < TOPK_) {
    float v = selv[threadIdx.x]; int n = seli[threadIdx.x];
    // mask * (align > 1e-9) * (align == column max over m); argmax = min m
    if (v > 1e-9f && v == colmax[(size_t)b * N_ + n])
      atomicMin(&tgtc[b * N_ + n], m);
  }
}

// Per anchor: -x*target term of BCE, box IoU loss, fg count.
__global__ __launch_bounds__(256) void k_anchor_loss(const float* __restrict__ ps,
                                                     const float* __restrict__ pb,
                                                     const float* __restrict__ gtb,
                                                     const int*   __restrict__ gtl,
                                                     const float* __restrict__ colmax,
                                                     const int*   __restrict__ tgtc,
                                                     float* __restrict__ accum) {
  __shared__ float sbuf[256];
  int i = blockIdx.x * 256 + threadIdx.x;
  float xts = 0.f, boxs = 0.f, fgs = 0.f;
  if (i < B_ * N_) {
    int b = i / N_;
    int cand = tgtc[i];
    bool fg = cand < M_;
    int tgt = fg ? cand : 0;
    int lbl = gtl[b * M_ + tgt];
    float ts = fg ? colmax[i] : 0.f;       // one_hot * (max_m * fg_mask)
    float x = ps[(size_t)i * NC_ + lbl];
    xts = x * ts;
    if (fg) {
      const float* bp = pb + (size_t)i * 4;
      const float* g  = gtb + ((size_t)b * M_ + tgt) * 4;
      float iw = fmaxf(fminf(bp[2], g[2]) - fmaxf(bp[0], g[0]), 0.f);
      float ih = fmaxf(fminf(bp[3], g[3]) - fmaxf(bp[1], g[1]), 0.f);
      float inter = iw * ih;
      float a1 = (bp[2] - bp[0]) * (bp[3] - bp[1]);
      float a2 = (g[2] - g[0]) * (g[3] - g[1]);
      float iou = inter / (a1 + a2 - inter + 1e-16f);
      boxs = 1.f - iou; fgs = 1.f;
    }
  }
  float s1 = block_sum(xts, sbuf);
  float s2 = block_sum(boxs, sbuf);
  float s3 = block_sum(fgs, sbuf);
  if (threadIdx.x == 0) {
    atomicAdd(&accum[1], s1);
    atomicAdd(&accum[2], s2);
    atomicAdd(&accum[3], s3);
  }
}

// Target-independent part of BCE: sum over ALL (B,N,NC).
__global__ __launch_bounds__(256) void k_softplus(const float* __restrict__ ps,
                                                  float* __restrict__ accum, int total) {
  __shared__ float sbuf[256];
  float s = 0.f;
  for (int i = blockIdx.x * 256 + threadIdx.x; i < total; i += gridDim.x * 256) {
    float x = ps[i];
    s += fmaxf(x, 0.f) + log1pf(__expf(-fabsf(x)));
  }
  s = block_sum(s, sbuf);
  if (threadIdx.x == 0) atomicAdd(&accum[0], s);
}

__global__ void k_final(const float* accum, float* out) {
  float sp = accum[0], xts = accum[1], box = accum[2], fg = accum[3];
  float ts_sum = fmaxf(fg, 1.f);
  float lcls = (sp - xts) / ts_sum;
  float lbox = (fg > 0.f) ? (box / fg) : 0.f;
  out[0] = lcls + 1.5f * lbox;
}

// ---------------------------------------------------------------- launch ----

extern "C" void kernel_launch(void* const* d_in, const int* in_sizes, int n_in,
                              void* d_out, int out_size, void* d_ws, size_t ws_size,
                              hipStream_t stream) {
  (void)in_sizes; (void)n_in; (void)out_size; (void)ws_size;
  const float* ps = (const float*)d_in[0];   // (B,N,NC)
  const float* pd = (const float*)d_in[1];   // (B,N,64)
  const int*   gl = (const int*)d_in[2];     // (B,M,1)
  const float* gb = (const float*)d_in[3];   // (B,M,4)
  const float* dw = (const float*)d_in[4];   // (16)
  float* out = (float*)d_out;

  float* pb     = (float*)d_ws;                      // B*N*4
  float* colmax = pb + (size_t)B_ * N_ * 4;          // B*N
  int*   tgtc   = (int*)(colmax + (size_t)B_ * N_);  // B*N
  float* accum  = (float*)(tgtc + (size_t)B_ * N_);  // 4

  k_init<<<1, 32, 0, stream>>>(accum);
  k_dfl_bbox<<<(B_ * N_) / 32, 256, 0, stream>>>(pd, dw, pb);     // exact: 268800/32
  k_colmax<<<(B_ * N_ + 255) / 256, 256, 0, stream>>>(ps, pb, gb, gl, colmax, tgtc);
  k_assign<<<B_ * M_, 256, 0, stream>>>(ps, pb, gb, gl, colmax, tgtc);
  k_anchor_loss<<<(B_ * N_ + 255) / 256, 256, 0, stream>>>(ps, pb, gb, gl, colmax, tgtc, accum);
  k_softplus<<<4096, 256, 0, stream>>>(ps, accum, B_ * N_ * NC_);
  k_final<<<1, 1, 0, stream>>>(accum, out);
}
